// GRUBlock_15367392985483
// MI455X (gfx1250) — matrix operand order repaired and checked by reading, not compiled, not run on hardware
//
#include <hip/hip_runtime.h>
#include <cstdint>

#define BB 16
#define LL 2048
#define CC 256
#define HH 256
#define GG 768            // 3*H
#define BL (BB*LL)        // 32768 rows

typedef __attribute__((ext_vector_type(16))) _Float16 v16h;
typedef __attribute__((ext_vector_type(8)))  float    v8f;
typedef __attribute__((ext_vector_type(4)))  unsigned int u32x4;
typedef __attribute__((ext_vector_type(8)))  int      i32x8;
typedef __attribute__((ext_vector_type(4)))  int      i32x4;

union HF { v16h v; uint4 q[2]; };

#if __has_builtin(__builtin_amdgcn_tensor_load_to_lds)
#define HAVE_TDM 1
#endif

// ---------------------------------------------------------------------------
// WMMA fragment loaders (layouts per CDNA5 ISA 7.12.2, wave32)
// ---------------------------------------------------------------------------

// B-matrix fragment (K=32 x N=16) from row-major f16 weights W[N_total][K]:
// lane&15 = n; lanes<16: K = kb+{0..7,16..23}; lanes>=16: K = kb+{8..15,24..31}
__device__ __forceinline__ v16h load_w_frag_h(const _Float16* __restrict__ W, int K,
                                              int colbase, int kbase) {
  int lane = threadIdx.x & 31;
  int n    = lane & 15;
  int ks   = kbase + ((lane & 16) ? 8 : 0);
  const _Float16* p = W + (size_t)(colbase + n) * (size_t)K + ks;
  HF f;
  f.q[0] = *reinterpret_cast<const uint4*>(p);
  f.q[1] = *reinterpret_cast<const uint4*>(p + 16);
  return f.v;
}

// A-matrix fragment (M=16 x K=32) from row-major f16 A[rows][K]:
// lane&15 = m; lanes<16: K = kb+{0..7,16..23}; lanes>=16: K = kb+{8..15,24..31}
__device__ __forceinline__ v16h load_a_frag(const _Float16* __restrict__ A, int K,
                                            int rowbase, int kbase) {
  int lane = threadIdx.x & 31;
  int m    = lane & 15;
  int sel  = (lane & 16) ? 8 : 0;
  const _Float16* p = A + (size_t)(rowbase + m) * (size_t)K + kbase + sel;
  HF f;
  f.q[0] = *reinterpret_cast<const uint4*>(p);       // K = kb+sel .. +7
  f.q[1] = *reinterpret_cast<const uint4*>(p + 16);  // K = kb+16+sel .. +7
  return f.v;
}

// ---------------------------------------------------------------------------
// TDM: DMA a (tile_rows x K) f16 tile out of a (tensor_rows x K) row-major
// tensor into LDS.  D# layout per CDNA5 ISA ch.8 (group0/group1).
// 6-arg builtin form (clang-23 / amdgpu-toolchain): (g0, g1, g2, g3, g4, cpol)
// ---------------------------------------------------------------------------
__device__ __forceinline__ void tdm_load_rows(unsigned int lds_off,
                                              const _Float16* gsrc, int K,
                                              int tensor_rows, int tile_rows) {
#ifdef HAVE_TDM
  unsigned long long ga = (unsigned long long)(uintptr_t)gsrc;
  u32x4 g0 = { 1u,                                   // count = 1 (valid D#)
               lds_off,                              // lds_addr
               (unsigned int)ga,                     // global_addr[31:0]
               (unsigned int)((ga >> 32) & 0x01FFFFFFu) | (2u << 30) }; // hi|type=2
  i32x8 g1 = { (int)(1u << 16),                                // data_size = 2B
               (int)((unsigned)(K & 0xFFFF) << 16),            // tensor_dim0 lo16
               (int)((unsigned)(tensor_rows & 0xFFFF) << 16),  // tensor_dim1 lo16
               (int)((unsigned)(K & 0xFFFF) << 16),            // tile_dim0
               tile_rows,                                      // tile_dim1
               K,                                              // dim0_stride lo32
               0, 0 };
  i32x4 z4 = { 0, 0, 0, 0 };
  i32x8 z8 = { 0, 0, 0, 0, 0, 0, 0, 0 };
  __builtin_amdgcn_tensor_load_to_lds(g0, g1, z4, z4, z8, 0);
#endif
}

// ---------------------------------------------------------------------------
// f32 -> f16 weight conversion (run once per weight matrix)
// ---------------------------------------------------------------------------
__global__ void __launch_bounds__(256) cvt_f32_f16(const float* __restrict__ s,
                                                   _Float16* __restrict__ d, int n) {
  int i = blockIdx.x * 256 + threadIdx.x;
  if (i < n) d[i] = (_Float16)s[i];
}

// ---------------------------------------------------------------------------
// Transpose + f16 convert: x (B,C,L) f32 -> xT (B,L,C) f16
// ---------------------------------------------------------------------------
__global__ void __launch_bounds__(256) xpose_kernel(const float* __restrict__ x,
                                                    _Float16* __restrict__ xT) {
  int i = blockIdx.x * 256 + threadIdx.x;           // over B*L*C = 2^23
  int c = i & (CC - 1);
  int l = (i >> 8) & (LL - 1);
  int b = i >> 19;
  xT[i] = (_Float16)x[((size_t)b * CC + c) * LL + l];
}

// ---------------------------------------------------------------------------
// Generic WMMA GEMM: out[row][col] = act( sum_k A[row][k]*W[col][k] + bias[col] )
// A: f16 row-major (BL x K).  W: f16 row-major (N x K).
// mode 0: store f16 row-major; mode 1: + ReLU, f16; mode 2: store f32 into
//         (B,C,L) layout (final output transpose).
// 8 waves / block; the 8 waves share one 16-row A tile (tilesN % 8 == 0),
// which is staged into LDS by the Tensor Data Mover.
// ---------------------------------------------------------------------------
__global__ void __launch_bounds__(256) wmma_gemm(const _Float16* __restrict__ A,
                                                 const _Float16* __restrict__ W,
                                                 const float* __restrict__ bias,
                                                 int N, int K, int mode,
                                                 _Float16* __restrict__ outH,
                                                 float* __restrict__ outF) {
  __shared__ __align__(16) _Float16 Atile[16 * 512];   // up to 16 KB

  const int tilesN  = N >> 4;
  const int tM      = (blockIdx.x * 8) / tilesN;        // uniform per block
  const int tN      = (blockIdx.x * 8) % tilesN + (threadIdx.x >> 5);
  const int lane    = threadIdx.x & 31;
  const int rowbase = tM << 4;
  const int colbase = tN << 4;

  // Stage the shared 16 x K A-tile into LDS (tile rows are contiguous).
#ifdef HAVE_TDM
  if ((threadIdx.x >> 5) == 0) {
    tdm_load_rows((unsigned int)(uintptr_t)(&Atile[0]),
                  A + (size_t)rowbase * K, K, BL, 16);
    __builtin_amdgcn_s_wait_tensorcnt(0);
  }
#else
  {
    const uint4* src = reinterpret_cast<const uint4*>(A + (size_t)rowbase * K);
    uint4*       dst = reinterpret_cast<uint4*>(Atile);
    const int nq = (16 * K) >> 3;
    for (int i = threadIdx.x; i < nq; i += 256) dst[i] = src[i];
  }
#endif
  __syncthreads();

  v8f acc = {};
  const int nkt = K >> 5;
  for (int kt = 0; kt < nkt; ++kt) {
    v16h a = load_a_frag(Atile, K, 0, kt * 32);        // ds_read_b128 x2
    v16h b = load_w_frag_h(W, K, colbase, kt * 32);    // global f16 x2
    acc = __builtin_amdgcn_wmma_f32_16x16x32_f16(false, a, false, b,
                                                 (short)0, acc, false, false);
  }

  const int   col = colbase + (lane & 15);
  const float bv  = bias ? bias[col] : 0.0f;
#pragma unroll
  for (int r = 0; r < 8; ++r) {
    const int   row = rowbase + ((lane & 16) ? (8 + r) : r);
    float v = acc[r] + bv;
    if (mode == 1) v = v > 0.0f ? v : 0.0f;
    if (mode == 2) {
      const int b_ = row >> 11;            // row = b*L + l, L = 2048
      const int l  = row & (LL - 1);
      outF[(size_t)b_ * CC * LL + (size_t)col * LL + l] = v;
    } else {
      outH[(size_t)row * N + col] = (_Float16)v;
    }
  }
}

// ---------------------------------------------------------------------------
// GRU recurrent scan. One block per direction (blockIdx.x: 0=fwd, 1=bwd).
// 768 threads = 24 waves; each wave owns 2 gate tiles (48 tiles = 768 gates)
// with its w_hh slice resident in VGPRs (2x8 v16h = 128 VGPRs) for all 2048
// steps. h lives in LDS (f16, WMMA-A layout-friendly); gh (16x768 f32) is
// exchanged through LDS each step.
// ---------------------------------------------------------------------------
__global__ void __launch_bounds__(768) gru_scan(const _Float16* __restrict__ gi_f,
                                                const _Float16* __restrict__ gi_b,
                                                const _Float16* __restrict__ whh_f,
                                                const _Float16* __restrict__ whh_b,
                                                const float* __restrict__ bhh_f,
                                                const float* __restrict__ bhh_b,
                                                _Float16* __restrict__ out) {
  __shared__ __align__(16) _Float16 hA[BB * HH];   // 8 KB hidden state (f16)
  __shared__ __align__(16) float    gh[BB * GG];   // 48 KB recurrent gate pre-acts
  __shared__ float                  bhh[GG];       // 3 KB recurrent bias

  const int dir  = blockIdx.x;
  const _Float16* gi   = dir ? gi_b  : gi_f;
  const _Float16* whh  = dir ? whh_b : whh_f;
  const float*    bhhg = dir ? bhh_b : bhh_f;
  const int       col0 = dir ? HH : 0;

  const int tid   = threadIdx.x;
  const int lane  = tid & 31;
  const int wave  = tid >> 5;
  const int gbase = wave * 32;                     // 2 x 16 gate columns

  for (int i = tid; i < BB * HH; i += 768) hA[i] = (_Float16)0.0f;
  for (int i = tid; i < GG;      i += 768) bhh[i] = bhhg[i];

  // Preload this wave's w_hh slice into registers (f16, two b128 per frag).
  v16h Breg[2][8];
#pragma unroll
  for (int t = 0; t < 2; ++t)
#pragma unroll
    for (int kt = 0; kt < 8; ++kt)
      Breg[t][kt] = load_w_frag_h(whh, HH, gbase + t * 16, kt * 32);

  __syncthreads();

  for (int step = 0; step < LL; ++step) {
    const int time = dir ? (LL - 1 - step) : step;

    // gh tile = h (16x256) @ w_hh_slice^T : 8 K-steps x 2 N-tiles of WMMA.
    v8f acc0 = {}, acc1 = {};
#pragma unroll
    for (int kt = 0; kt < 8; ++kt) {
      const int sel = (lane & 16) ? 8 : 0;
      const _Float16* p = hA + (lane & 15) * HH + kt * 32 + sel;
      HF f;
      f.q[0] = *reinterpret_cast<const uint4*>(p);
      f.q[1] = *reinterpret_cast<const uint4*>(p + 16);
      acc0 = __builtin_amdgcn_wmma_f32_16x16x32_f16(false, f.v, false, Breg[0][kt],
                                                    (short)0, acc0, false, false);
      acc1 = __builtin_amdgcn_wmma_f32_16x16x32_f16(false, f.v, false, Breg[1][kt],
                                                    (short)0, acc1, false, false);
    }

    // Publish gh to LDS (C/D layout: vgpr r -> M = r or 8+r, lane&15 -> N).
#pragma unroll
    for (int r = 0; r < 8; ++r) {
      const int m = (lane & 16) ? (8 + r) : r;
      const int n = lane & 15;
      gh[m * GG + gbase + n]      = acc0[r];
      gh[m * GG + gbase + 16 + n] = acc1[r];
    }
    __syncthreads();

    // Elementwise GRU cell for 16x256 hidden units.
    for (int s = tid; s < BB * HH; s += 768) {
      const int b = s >> 8, j = s & 255;
      const float*    ghrow = gh + b * GG;
      const _Float16* gip   = gi + ((size_t)b * LL + time) * GG;
      const float hprev  = (float)hA[b * HH + j];
      const float pr     = (float)gip[j]      + ghrow[j]      + bhh[j];
      const float pz     = (float)gip[HH + j] + ghrow[HH + j] + bhh[HH + j];
      const float hn_pre = ghrow[2 * HH + j] + bhh[2 * HH + j];
      const float r = 1.0f / (1.0f + __expf(-pr));
      const float z = 1.0f / (1.0f + __expf(-pz));
      const float n = tanhf((float)gip[2 * HH + j] + r * hn_pre);
      const float hnew = (1.0f - z) * n + z * hprev;
      hA[b * HH + j] = (_Float16)hnew;
      out[((size_t)b * LL + time) * 512 + col0 + j] = (_Float16)hnew;
    }
    __syncthreads();
  }
}

// ---------------------------------------------------------------------------
// Direction sum + LayerNorm: in (B,L,512) f16 -> out (B,L,256) f16.
// One wave32 per row; shfl_xor reductions.
// ---------------------------------------------------------------------------
__global__ void __launch_bounds__(256) sum_ln(const _Float16* __restrict__ in,
                                              const float* __restrict__ g,
                                              const float* __restrict__ beta,
                                              _Float16* __restrict__ out) {
  const int wave = threadIdx.x >> 5;
  const int lane = threadIdx.x & 31;
  const int row  = blockIdx.x * 8 + wave;          // < BL
  const _Float16* p = in + (size_t)row * 512;

  float v[8], s = 0.0f, s2 = 0.0f;
#pragma unroll
  for (int j = 0; j < 8; ++j) {
    const int c = j * 32 + lane;
    v[j] = (float)p[c] + (float)p[256 + c];        // fwd + bwd
    s  += v[j];
    s2 += v[j] * v[j];
  }
#pragma unroll
  for (int m = 16; m >= 1; m >>= 1) {
    s  += __shfl_xor(s,  m, 32);
    s2 += __shfl_xor(s2, m, 32);
  }
  const float mu  = s * (1.0f / 256.0f);
  const float var = s2 * (1.0f / 256.0f) - mu * mu;
  const float rs  = rsqrtf(var + 1e-5f);
#pragma unroll
  for (int j = 0; j < 8; ++j) {
    const int c = j * 32 + lane;
    out[(size_t)row * 256 + c] = (_Float16)((v[j] - mu) * rs * g[c] + beta[c]);
  }
}

// ---------------------------------------------------------------------------
// Host-side launch.  Workspace layout (assumes ws_size >= ~165 MB):
//   [0,48M)     gi_f f16 (B,L,768)  -> reused as ln_out f16 (B,L,256)
//   [48M,96M)   gi_b f16 (B,L,768)  -> reused as relu  f16 (B,L,512)
//   [96M,128M)  bufA f16 (B,L,512)  (xT then layer-1 output)
//   [128M,160M) bufB f16 (B,L,512)  (layer-0 output / layer-1 input)
//   [160M,..)   f16 weight arena (~4.4 MB)
// ---------------------------------------------------------------------------
extern "C" void kernel_launch(void* const* d_in, const int* in_sizes, int n_in,
                              void* d_out, int out_size, void* d_ws, size_t ws_size,
                              hipStream_t stream) {
  const float* x      = (const float*)d_in[0];
  const float* w_ih0f = (const float*)d_in[1];
  const float* w_hh0f = (const float*)d_in[2];
  const float* b_ih0f = (const float*)d_in[3];
  const float* b_hh0f = (const float*)d_in[4];
  const float* w_ih0b = (const float*)d_in[5];
  const float* w_hh0b = (const float*)d_in[6];
  const float* b_ih0b = (const float*)d_in[7];
  const float* b_hh0b = (const float*)d_in[8];
  const float* w_ih1f = (const float*)d_in[9];
  const float* w_hh1f = (const float*)d_in[10];
  const float* b_ih1f = (const float*)d_in[11];
  const float* b_hh1f = (const float*)d_in[12];
  const float* w_ih1b = (const float*)d_in[13];
  const float* w_hh1b = (const float*)d_in[14];
  const float* b_ih1b = (const float*)d_in[15];
  const float* b_hh1b = (const float*)d_in[16];
  const float* ln_g   = (const float*)d_in[17];
  const float* ln_b   = (const float*)d_in[18];
  const float* w1     = (const float*)d_in[19];
  const float* b1     = (const float*)d_in[20];
  const float* w2     = (const float*)d_in[21];
  const float* b2     = (const float*)d_in[22];

  char* ws = (char*)d_ws;
  const size_t GI_BYTES  = (size_t)BL * GG * sizeof(_Float16);   // 48 MB
  const size_t BUF_BYTES = (size_t)BL * 512 * sizeof(_Float16);  // 32 MB
  _Float16* gi_f = (_Float16*)(ws);
  _Float16* gi_b = (_Float16*)(ws + GI_BYTES);
  _Float16* bufA = (_Float16*)(ws + 2 * GI_BYTES);
  _Float16* bufB = (_Float16*)(ws + 2 * GI_BYTES + BUF_BYTES);
  _Float16* lnb  = gi_f;   // reuse after scans done
  _Float16* relu = gi_b;   // reuse after scans done

  // f16 weight arena
  _Float16* wh = (_Float16*)(ws + 2 * GI_BYTES + 2 * BUF_BYTES);
  struct WEnt { const float* src; _Float16* dst; int n; };
  _Float16* p = wh;
  WEnt ents[10];
  const float* srcs[10] = { w_ih0f, w_ih0b, w_ih1f, w_ih1b,
                            w_hh0f, w_hh0b, w_hh1f, w_hh1b, w1, w2 };
  const int    ns[10]   = { GG*256, GG*256, GG*512, GG*512,
                            GG*256, GG*256, GG*256, GG*256, 512*256, 256*512 };
  for (int i = 0; i < 10; ++i) { ents[i] = { srcs[i], p, ns[i] }; p += ns[i]; }
  for (int i = 0; i < 10; ++i)
    cvt_f32_f16<<<(ents[i].n + 255) / 256, 256, 0, stream>>>(ents[i].src, ents[i].dst, ents[i].n);
  _Float16* wih0f_h = ents[0].dst; _Float16* wih0b_h = ents[1].dst;
  _Float16* wih1f_h = ents[2].dst; _Float16* wih1b_h = ents[3].dst;
  _Float16* whh0f_h = ents[4].dst; _Float16* whh0b_h = ents[5].dst;
  _Float16* whh1f_h = ents[6].dst; _Float16* whh1b_h = ents[7].dst;
  _Float16* w1_h    = ents[8].dst; _Float16* w2_h    = ents[9].dst;

  const int gridGi = (BL / 16) * (GG / 16) / 8;    // 12288
  const int gridF1 = (BL / 16) * (512 / 16) / 8;   //  8192
  const int gridF2 = (BL / 16) * (256 / 16) / 8;   //  4096

  // 1) x (B,C,L) f32 -> bufA (B,L,256) f16
  xpose_kernel<<<BL * CC / 256, 256, 0, stream>>>(x, bufA);

  // 2-3) layer-0 input gates: gi = xT @ w_ih^T + b_ih
  wmma_gemm<<<gridGi, 256, 0, stream>>>(bufA, wih0f_h, b_ih0f, GG, 256, 0, gi_f, nullptr);
  wmma_gemm<<<gridGi, 256, 0, stream>>>(bufA, wih0b_h, b_ih0b, GG, 256, 0, gi_b, nullptr);

  // 4) layer-0 bidirectional recurrent scan -> bufB (B,L,512)
  gru_scan<<<2, 768, 0, stream>>>(gi_f, gi_b, whh0f_h, whh0b_h, b_hh0f, b_hh0b, bufB);

  // 5-6) layer-1 input gates (K = 512)
  wmma_gemm<<<gridGi, 256, 0, stream>>>(bufB, wih1f_h, b_ih1f, GG, 512, 0, gi_f, nullptr);
  wmma_gemm<<<gridGi, 256, 0, stream>>>(bufB, wih1b_h, b_ih1b, GG, 512, 0, gi_b, nullptr);

  // 7) layer-1 scan -> bufA (B,L,512)
  gru_scan<<<2, 768, 0, stream>>>(gi_f, gi_b, whh1f_h, whh1b_h, b_hh1f, b_hh1b, bufA);

  // 8) fwd+bwd sum and LayerNorm -> lnb (B,L,256) f16
  sum_ln<<<BL / 8, 256, 0, stream>>>(bufA, ln_g, ln_b, lnb);

  // 9) FFN GEMM1 + ReLU: (BL,256) @ w1^T -> relu (BL,512) f16
  wmma_gemm<<<gridF1, 256, 0, stream>>>(lnb, w1_h, b1, 512, 256, 1, relu, nullptr);

  // 10) FFN GEMM2: (BL,512) @ w2^T -> d_out f32 stored as (B,C,L)
  wmma_gemm<<<gridF2, 256, 0, stream>>>(relu, w2_h, b2, 256, 512, 2, nullptr, (float*)d_out);
}
